// RPN_90254442758559
// MI455X (gfx1250) — matrix operand order repaired and checked by reading, compiled, not run on hardware
//
#include <hip/hip_runtime.h>
#include <hip/hip_bf16.h>

typedef __attribute__((ext_vector_type(16))) __bf16 v16bf;
typedef __attribute__((ext_vector_type(8)))  float  v8f;

#define R_TOTAL 327360
#define NUM_K_STEPS 72          // 2304 / 32
#define AS_STRIDE 40            // halves per A row (80 B: b128-aligned stores)
#define BS_STRIDE 36            // halves per B row (72 B: b64-aligned reads)
#define TS_STRIDE 257           // padded floats per t row (256 + 1)
#define SMEM_BYTES (32 * TS_STRIDE * 4)   // 32896 > 256*40*2 + 64*36*2 = 25088

// ---------------------------------------------------------------------------
// fp32 -> bf16 (RNE) elementwise conversion
// ---------------------------------------------------------------------------
__global__ void rpn_cvt_bf16(const float* __restrict__ in,
                             unsigned short* __restrict__ out, int n) {
    int i = blockIdx.x * blockDim.x + threadIdx.x;
    if (i < n) {
        unsigned int u = __float_as_uint(in[i]);
        u += 0x7FFFu + ((u >> 16) & 1u);
        out[i] = (unsigned short)(u >> 16);
    }
}

// packed (ci,kh,kw) per flattened k: avoids /9 %9 /3 in the hot gather
__global__ void rpn_make_ktab(unsigned* __restrict__ ktab) {
    int k = blockIdx.x * 256 + threadIdx.x;
    if (k < 2304) {
        int ci = k / 9, rem = k - ci * 9, kh = rem / 3, kw = rem - kh * 3;
        ktab[k] = ((unsigned)ci << 4) | ((unsigned)kh << 2) | (unsigned)kw;
    }
}

__global__ void rpn_init_keys(unsigned long long* keys) {
    if (threadIdx.x < 64) keys[threadIdx.x] = 0ull;
}

// ---------------------------------------------------------------------------
// Anchor labeling: per-anchor max-IoU / argmax vs 32 GT boxes, per-gt best
// anchor reduced through LDS atomicMax of packed (iou_bits << 32 | ~r) keys.
// ---------------------------------------------------------------------------
__global__ __launch_bounds__(256) void rpn_label_kernel(
    const float* __restrict__ gt,      // [2][32][4] xywh
    float* __restrict__ labels,        // [2][R]
    float* __restrict__ matched,       // [2][R][4]
    unsigned long long* __restrict__ keys) {   // [2][32]
    __shared__ float gx[32][4];
    __shared__ unsigned long long skey[32];
    const int img = blockIdx.y;
    const int tid = threadIdx.x;
    if (tid < 32) {
        float x = gt[(img * 32 + tid) * 4 + 0];
        float y = gt[(img * 32 + tid) * 4 + 1];
        float w = gt[(img * 32 + tid) * 4 + 2];
        float h = gt[(img * 32 + tid) * 4 + 3];
        gx[tid][0] = x; gx[tid][1] = y; gx[tid][2] = x + w; gx[tid][3] = y + h;
        skey[tid] = 0ull;
    }
    __syncthreads();

    const int r = blockIdx.x * 256 + tid;
    const bool valid = (r < R_TOTAL);
    float ax1 = 0.f, ay1 = 0.f, ax2 = 0.f, ay2 = 0.f;
    if (valid) {
        int lvl = 0, rl = r;
        if (r >= 326400)      { lvl = 4; rl = r - 326400; }
        else if (r >= 322560) { lvl = 3; rl = r - 322560; }
        else if (r >= 307200) { lvl = 2; rl = r - 307200; }
        else if (r >= 245760) { lvl = 1; rl = r - 245760; }
        const int gs = 7 - lvl;                 // log2(g)
        const float stride = 4.0f * (float)(1 << lvl);
        const int a = rl >> (2 * gs);
        const int p = rl & ((1 << (2 * gs)) - 1);
        const int ii = p >> gs, jj = p & ((1 << gs) - 1);
        const float cx = (float)ii * stride, cy = (float)jj * stride;
        const float S[5] = {32.f, 64.f, 128.f, 256.f, 512.f};
        const float s = S[a / 3];
        const int ri = a % 3;
        const float rs = (ri == 0) ? 0.70710678118654752f
                       : (ri == 1) ? 1.0f : 1.41421356237309515f;
        const float wa = s * rs, ha = s / rs;
        ax1 = cx - 0.5f * wa; ay1 = cy - 0.5f * ha;
        ax2 = cx + 0.5f * wa; ay2 = cy + 0.5f * ha;
    }
    const float areaA = (ax2 - ax1) * (ay2 - ay1);

    float bestIou = -1.0f;
    int bestIdx = 0;
    for (int n = 0; n < 32; ++n) {
        const float bx1 = gx[n][0], by1 = gx[n][1], bx2 = gx[n][2], by2 = gx[n][3];
        const float iw = fmaxf(fminf(ax2, bx2) - fmaxf(ax1, bx1), 0.f);
        const float ih = fmaxf(fminf(ay2, by2) - fmaxf(ay1, by1), 0.f);
        const float inter = iw * ih;
        const float areaB = (bx2 - bx1) * (by2 - by1);
        const float iou = inter / (areaA + areaB - inter);
        if (valid) {
            if (iou > bestIou) { bestIou = iou; bestIdx = n; }  // first-max wins
            unsigned long long key =
                ((unsigned long long)__float_as_uint(iou) << 32) |
                (unsigned long long)(0xFFFFFFFFu - (unsigned)r);
            atomicMax(&skey[n], key);
        }
    }
    if (valid) {
        const float lab = (bestIou < 0.3f) ? 0.0f : ((bestIou > 0.7f) ? 1.0f : -1.0f);
        labels[(long long)img * R_TOTAL + r] = lab;
        const long long mi = ((long long)img * R_TOTAL + r) * 4;
        matched[mi + 0] = gx[bestIdx][0];
        matched[mi + 1] = gx[bestIdx][1];
        matched[mi + 2] = gx[bestIdx][2];
        matched[mi + 3] = gx[bestIdx][3];
    }
    __syncthreads();
    if (tid < 32) atomicMax(&keys[img * 32 + tid], skey[tid]);
}

__global__ void rpn_fixup_kernel(const unsigned long long* __restrict__ keys,
                                 float* __restrict__ labels) {
    const int t = threadIdx.x;
    if (t < 64) {
        const int img = t >> 5, n = t & 31;
        const unsigned long long key = keys[img * 32 + n];
        const unsigned r = 0xFFFFFFFFu - (unsigned)(key & 0xFFFFFFFFull);
        if (r < (unsigned)R_TOTAL) {
            float* lab = labels + (long long)img * R_TOTAL + r;
            if (*lab != 0.0f) *lab = 1.0f;   // neg (score<0.3) overrides forced pos
        }
    }
}

// ---------------------------------------------------------------------------
// Fused 3x3-conv(256->256, bf16 WMMA, fp32 accum) + ReLU + dual 1x1 heads.
// Block: 256 threads (8 waves, 4M x 2N), tile = 256 co x 64 positions.
// A staged as 4x b128 load + 4x b128 LDS store per thread (one row each);
// 4 A-frags reused across 2 N-frags -> 8 WMMA per K-step via ds_load_b64.
// ---------------------------------------------------------------------------
__global__ __launch_bounds__(256) void rpn_head_kernel(
    const unsigned short* __restrict__ Aw,    // [256][2304] bf16 weights
    const unsigned short* __restrict__ Xbf,   // [2][256][g][g] bf16 feats
    const unsigned* __restrict__ ktab,        // [2304] packed (ci,kh,kw)
    const float* __restrict__ conv_b,         // [256]
    const float* __restrict__ cls_w,          // [15][256]
    const float* __restrict__ cls_b,          // [15]
    const float* __restrict__ box_w,          // [60][256]
    const float* __restrict__ box_b,          // [60]
    float* __restrict__ preds,                // [2][R][5]
    int gshift, int lvlBase) {
    __shared__ __align__(16) unsigned char smem[SMEM_BYTES];
    unsigned short* As = (unsigned short*)smem;                          // [256][40]
    unsigned short* Bs = (unsigned short*)(smem + 256 * AS_STRIDE * 2);  // [64][36]
    float* tS = (float*)smem;                                            // [32][257]

    const int tid   = threadIdx.x;
    const int lane  = tid & 31;
    const int wave  = tid >> 5;
    const int mBase = (wave & 3) * 64;
    const int nBase = (wave >> 2) * 16;
    const int img   = blockIdx.y;
    const int n0    = blockIdx.x * 64;
    const int g     = 1 << gshift;
    const int gg    = 1 << (2 * gshift);

    const unsigned short* X = Xbf + ((size_t)img << (8 + 2 * gshift));

    // loop-invariant pieces of the B gather (kk is the same every i)
    const int bn0 = tid >> 5;            // first of 8 strided n slots
    const int kk  = tid & 31;

    v8f acc[4][2] = {};

    for (int ks = 0; ks < NUM_K_STEPS; ++ks) {
        const int k0 = ks * 32;
        // ---- stage A tile (256x32 bf16): one contiguous 64 B row per thread
        {
            const unsigned short* src = Aw + (size_t)tid * 2304 + k0;
            const uint4 t0 = *(const uint4*)(src + 0);
            const uint4 t1 = *(const uint4*)(src + 8);
            const uint4 t2 = *(const uint4*)(src + 16);
            const uint4 t3 = *(const uint4*)(src + 24);
            unsigned short* dst = As + tid * AS_STRIDE;
            *(uint4*)(dst + 0)  = t0;
            *(uint4*)(dst + 8)  = t1;
            *(uint4*)(dst + 16) = t2;
            *(uint4*)(dst + 24) = t3;
        }
        // prefetch next K-slice of weights (global_prefetch_b8)
        if (ks + 1 < NUM_K_STEPS)
            __builtin_prefetch(Aw + (size_t)tid * 2304 + k0 + 32, 0, 1);

        // ---- stage B tile (64 pos x 32 k, implicit-GEMM gather)
        {
            const unsigned pk = ktab[k0 + kk];
            const int ci = (int)(pk >> 4);
            const int kh = (int)((pk >> 2) & 3u);
            const int kw = (int)(pk & 3u);
            const unsigned short* Xc = X + ((size_t)ci << (2 * gshift));
#pragma unroll
            for (int i = 0; i < 8; ++i) {
                const int n = bn0 + i * 8;
                const int p = n0 + n;
                unsigned short v = 0;
                if (p < gg) {
                    const int ii = p >> gshift, jj = p & (g - 1);
                    const int si = ii + kh - 1, sj = jj + kw - 1;
                    if ((unsigned)si < (unsigned)g && (unsigned)sj < (unsigned)g)
                        v = Xc[(si << gshift) + sj];
                }
                Bs[n * BS_STRIDE + kk] = v;
            }
        }
        __syncthreads();

        // ---- 2 B fragments (32x16 bf16 each): 4 ds_load_b64 per frag
        union Frag { v16bf v; unsigned long long q[4]; };
        Frag bfr[2];
#pragma unroll
        for (int nf = 0; nf < 2; ++nf) {
            const int n = nBase + nf * 32 + (lane & 15);
#pragma unroll
            for (int j = 0; j < 4; ++j) {
                const int qb = (lane < 16) ? (4 * j) : (16 + 4 * j);
                bfr[nf].q[j] =
                    *(const unsigned long long*)(Bs + n * BS_STRIDE + qb);
            }
        }
        // ---- 4 A fragments, each reused for both N-frags -> 8 WMMA
#pragma unroll
        for (int mf = 0; mf < 4; ++mf) {
            Frag af;
            const int row = mBase + mf * 16 + (lane & 15);
#pragma unroll
            for (int j = 0; j < 4; ++j) {
                const int qb = (lane < 16) ? (4 * j) : (16 + 4 * j);
                af.q[j] = *(const unsigned long long*)(As + row * AS_STRIDE + qb);
            }
            acc[mf][0] = __builtin_amdgcn_wmma_f32_16x16x32_bf16(
                false, af.v, false, bfr[0].v, (short)0, acc[mf][0], false, false);
            acc[mf][1] = __builtin_amdgcn_wmma_f32_16x16x32_bf16(
                false, af.v, false, bfr[1].v, (short)0, acc[mf][1], false, false);
        }
        __syncthreads();
    }

    // ---- two 32-column halves: bias+ReLU -> t LDS -> fused 1x1 heads
    for (int half = 0; half < 2; ++half) {
        __syncthreads();
#pragma unroll
        for (int mf = 0; mf < 4; ++mf) {
#pragma unroll
            for (int rr = 0; rr < 8; ++rr) {
                const int M = mBase + mf * 16 + rr + ((lane < 16) ? 0 : 8);
                const int N = nBase + (lane & 15);      // local 0..31
                const float v = acc[mf][half][rr] + conv_b[M];
                tS[N * TS_STRIDE + M] = fmaxf(v, 0.0f);
            }
        }
        __syncthreads();

        // 15 cls + 60 box channels over this half's 32 positions
        for (int widx = tid; widx < 75 * 32; widx += 256) {
            const int q = widx >> 5;       // wave-uniform channel
            const int n = widx & 31;
            const int p = n0 + half * 32 + n;
            if (p >= gg) continue;
            const float* wv;
            float bias;
            long long outIdx;
            if (q < 15) {
                wv = cls_w + q * 256;
                bias = cls_b[q];
                outIdx = ((long long)img * R_TOTAL + lvlBase
                          + (long long)q * gg + p) * 5;
            } else {
                const int qq = q - 15;
                const int a = qq >> 2, c = qq & 3;
                wv = box_w + qq * 256;
                bias = box_b[qq];
                outIdx = ((long long)img * R_TOTAL + lvlBase
                          + (long long)a * gg + p) * 5 + 1 + c;
            }
            float s = bias;
            const float* trow = tS + n * TS_STRIDE;
#pragma unroll 8
            for (int co = 0; co < 256; ++co) s += wv[co] * trow[co];
            preds[outIdx] = s;
        }
    }
}

// ---------------------------------------------------------------------------
extern "C" void kernel_launch(void* const* d_in, const int* in_sizes, int n_in,
                              void* d_out, int out_size, void* d_ws, size_t ws_size,
                              hipStream_t stream) {
    (void)in_sizes; (void)n_in; (void)out_size; (void)ws_size;
    // inputs: 0 images, 1..5 feat0..feat4, 6 gt_boxes, 7 conv_w, 8 conv_b,
    //         9 cls_w, 10 cls_b, 11 box_w, 12 box_b
    const float* gt_boxes = (const float*)d_in[6];
    const float* conv_w   = (const float*)d_in[7];
    const float* conv_b   = (const float*)d_in[8];
    const float* cls_w    = (const float*)d_in[9];
    const float* cls_b    = (const float*)d_in[10];
    const float* box_w    = (const float*)d_in[11];
    const float* box_b    = (const float*)d_in[12];

    // workspace layout (bytes):
    //   [0)          bf16 weights   1,179,648
    //   [1,179,648)  bf16 feats    22,347,776
    //   [23,527,424) ktab           9,216
    //   [23,536,640) 64 x u64 per-gt keys
    unsigned short*     Aw     = (unsigned short*)d_ws;
    unsigned short*     featbf = (unsigned short*)((char*)d_ws + 1179648);
    unsigned*           ktab   = (unsigned*)((char*)d_ws + 23527424);
    unsigned long long* keys   = (unsigned long long*)((char*)d_ws + 23536640);

    float* preds   = (float*)d_out;            // [2][R][5]
    float* labels  = preds + 2 * R_TOTAL * 5;  // [2][R]
    float* matched = labels + 2 * R_TOTAL;     // [2][R][4]

    static const int    gsh[5]     = {7, 6, 5, 4, 3};
    static const size_t foff[5]    = {0, 8388608, 10485760, 11010048, 11141120}; // halves
    static const int    lvlBase[5] = {0, 245760, 307200, 322560, 326400};

    rpn_cvt_bf16<<<(589824 + 255) / 256, 256, 0, stream>>>(conv_w, Aw, 589824);
    for (int l = 0; l < 5; ++l) {
        const int n = 2 * 256 * (1 << (2 * gsh[l]));
        rpn_cvt_bf16<<<(n + 255) / 256, 256, 0, stream>>>(
            (const float*)d_in[1 + l], featbf + foff[l], n);
    }
    rpn_make_ktab<<<9, 256, 0, stream>>>(ktab);

    rpn_init_keys<<<1, 64, 0, stream>>>(keys);
    rpn_label_kernel<<<dim3((R_TOTAL + 255) / 256, 2), 256, 0, stream>>>(
        gt_boxes, labels, matched, keys);
    rpn_fixup_kernel<<<1, 64, 0, stream>>>(keys, labels);

    for (int l = 0; l < 5; ++l) {
        const int gg = 1 << (2 * gsh[l]);
        rpn_head_kernel<<<dim3((gg + 63) / 64, 2), 256, 0, stream>>>(
            Aw, featbf + foff[l], ktab, conv_b, cls_w, cls_b, box_w, box_b,
            preds, gsh[l], lvlBase[l]);
    }
}
